// ScaledDotProductAttention_16698832847503
// MI455X (gfx1250) — matrix operand order repaired and checked by reading, not compiled, not run on hardware
//
#include <hip/hip_runtime.h>

typedef __attribute__((ext_vector_type(16))) _Float16 v16h;
typedef __attribute__((ext_vector_type(8)))  _Float16 v8h;
typedef __attribute__((ext_vector_type(8)))  float    v8f;

#define NKEYS 1024
#define DKDIM 64
#define DVDIM 64

// 1024 blocks = 128 (b,h) x 8 q-chunks of 128 rows; 8 waves/block, 16 q-rows/wave.
__global__ __launch_bounds__(256, 2)
void sdpa_wmma_kernel(const float* __restrict__ Qg, const float* __restrict__ Kg,
                      const float* __restrict__ Vg, const float* __restrict__ Mg,
                      float* __restrict__ Octx, float* __restrict__ Oatt)
{
    __shared__ __align__(32) _Float16 Kt[64][64];        // K tile, row-major [key][d]
    __shared__ __align__(32) _Float16 Vt[64][64];        // V tile, TRANSPOSED [dv][key]
    __shared__ __align__(32) _Float16 Sw[8][16][64];     // per-wave normalized P tile

    const int tid  = threadIdx.x;
    const int wv   = tid >> 5;      // wave id 0..7
    const int lane = tid & 31;
    const int hh   = lane >> 4;     // half-wave 0/1
    const int l16  = lane & 15;

    const int bh    = blockIdx.x >> 3;
    const int qc    = blockIdx.x & 7;
    const int qrow0 = qc * 128 + wv * 16;

    const float* Qb = Qg + (size_t)bh * NKEYS * DKDIM;
    const float* Kb = Kg + (size_t)bh * NKEYS * DKDIM;
    const float* Vb = Vg + (size_t)bh * NKEYS * DVDIM;
    const float* Mb = Mg + (size_t)bh * NKEYS;
    float* Ob = Octx + (size_t)bh * NKEYS * DVDIM;
    float* Wb = Oatt + (size_t)bh * NKEYS * NKEYS;

    // ---- Q A-fragments (16x64 f16, two 16x32 k-steps), kept in VGPRs all kernel ----
    // A operand in this code's contiguous k order (not AMD's table order): lane holds row M=l16; elems 0..15 = K in [16h,16h+16).
    v16h qa[2];
    {
        const float* qrow = Qb + (size_t)(qrow0 + l16) * DKDIM;
        const int kb = hh * 16;
#pragma unroll
        for (int s = 0; s < 2; ++s) {
            v16h a;
#pragma unroll
            for (int e = 0; e < 8; ++e) {
                a[e]     = (_Float16)qrow[s * 32 + kb + e];
                a[e + 8] = (_Float16)qrow[s * 32 + kb + 8 + e];
            }
            qa[s] = a;
        }
    }

    float rs[8];
#pragma unroll
    for (int v = 0; v < 8; ++v) rs[v] = 0.0f;

    // ================= pass 1: masked-exp row sums (no stores) =================
    for (int kt = 0; kt < NKEYS / 64; ++kt) {
        __syncthreads();
#pragma unroll
        for (int i = 0; i < 16; ++i) {
            int e = i * 256 + tid;
            int r = e >> 6, c = e & 63;
            Kt[r][c] = (_Float16)Kb[(size_t)(kt * 64 + r) * DKDIM + c];
        }
        if (kt + 1 < NKEYS / 64)
            __builtin_prefetch(&Kb[(size_t)((kt + 1) * 64 + (tid >> 2)) * DKDIM], 0, 1);
        __syncthreads();

#pragma unroll
        for (int j = 0; j < 4; ++j) {
            // B = K^T: lane holds score-column n=l16 (key j*16+l16); elem e = d-index
            // 16*hh + e  -> contiguous 16 halfs of K row -> single 32B LDS read.
            v8f acc = {};
#pragma unroll
            for (int s = 0; s < 2; ++s) {
                v16h b = *(const v16h*)&Kt[j * 16 + l16][s * 32 + hh * 16];
                acc = __builtin_amdgcn_wmma_f32_16x16x32_f16(false, qa[s], false, b,
                                                             (short)0, acc, false, false);
            }
            const float m = Mb[kt * 64 + j * 16 + l16];
#pragma unroll
            for (int v = 0; v < 8; ++v)
                rs[v] += __expf(acc[v] * 0.125f) * m;   // 1/sqrt(64) = 0.125
        }
    }

    // cross-lane reduce within 16-lane halves; result lands exactly in the
    // (lane,vgpr) slot that pass-2 D-fragment row v+8*hh needs.
#pragma unroll
    for (int v = 0; v < 8; ++v) {
        float s = rs[v];
        s += __shfl_xor(s, 1, 32);
        s += __shfl_xor(s, 2, 32);
        s += __shfl_xor(s, 4, 32);
        s += __shfl_xor(s, 8, 32);
        rs[v] = 1.0f / (s + 1e-8f);
    }

    v8f ctx[4];
#pragma unroll
    for (int j = 0; j < 4; ++j) ctx[j] = v8f{};

    // ===== pass 2: recompute scores, normalize, store attn_w, accumulate P@V =====
    for (int kt = 0; kt < NKEYS / 64; ++kt) {
        __syncthreads();
#pragma unroll
        for (int i = 0; i < 16; ++i) {
            int e = i * 256 + tid;
            int r = e >> 6, c = e & 63;
            Kt[r][c] = (_Float16)Kb[(size_t)(kt * 64 + r) * DKDIM + c];
            Vt[c][r] = (_Float16)Vb[(size_t)(kt * 64 + r) * DVDIM + c];  // transpose
        }
        __syncthreads();

#pragma unroll
        for (int j = 0; j < 4; ++j) {
            v8f acc = {};
#pragma unroll
            for (int s = 0; s < 2; ++s) {
                v16h b = *(const v16h*)&Kt[j * 16 + l16][s * 32 + hh * 16];
                acc = __builtin_amdgcn_wmma_f32_16x16x32_f16(false, qa[s], false, b,
                                                             (short)0, acc, false, false);
            }
            const float m   = Mb[kt * 64 + j * 16 + l16];
            const int  key0 = kt * 64 + j * 16 + l16;
#pragma unroll
            for (int v = 0; v < 8; ++v) {
                // D layout: row = v + 8*hh, col(key) = l16
                float w = __expf(acc[v] * 0.125f) * m * rs[v];
                Wb[(size_t)(qrow0 + v + 8 * hh) * NKEYS + key0] = w;   // attn_w out
                Sw[wv][v + 8 * hh][j * 16 + l16] = (_Float16)w;        // P tile f16
            }
        }

        // context += P(16x64) @ V(64x64), two 16x16x32 k-steps over keys
#pragma unroll
        for (int s = 0; s < 2; ++s) {
            v8h plo = *(const v8h*)&Sw[wv][l16][s * 32 + hh * 16];
            v8h phi = *(const v8h*)&Sw[wv][l16][s * 32 + hh * 16 + 8];
            v16h pa;
#pragma unroll
            for (int e = 0; e < 8; ++e) { pa[e] = plo[e]; pa[e + 8] = phi[e]; }
#pragma unroll
            for (int j2 = 0; j2 < 4; ++j2) {
                // B = V: lane holds dv-column j2*16+l16; elem e = key 16*hh+e
                // -> contiguous in transposed Vt -> single 32B LDS read.
                v16h b = *(const v16h*)&Vt[j2 * 16 + l16][s * 32 + hh * 16];
                ctx[j2] = __builtin_amdgcn_wmma_f32_16x16x32_f16(false, pa, false, b,
                                                                 (short)0, ctx[j2], false, false);
            }
        }
    }

    // store context: D layout row = v + 8*hh, col = j2*16 + l16
#pragma unroll
    for (int j2 = 0; j2 < 4; ++j2)
#pragma unroll
        for (int v = 0; v < 8; ++v)
            Ob[(size_t)(qrow0 + v + 8 * hh) * DVDIM + j2 * 16 + l16] = ctx[j2][v];
}

extern "C" void kernel_launch(void* const* d_in, const int* in_sizes, int n_in,
                              void* d_out, int out_size, void* d_ws, size_t ws_size,
                              hipStream_t stream) {
    (void)in_sizes; (void)n_in; (void)d_ws; (void)ws_size; (void)out_size;
    const float* Q = (const float*)d_in[0];
    const float* K = (const float*)d_in[1];
    const float* V = (const float*)d_in[2];
    const float* M = (const float*)d_in[3];
    float* out  = (float*)d_out;
    float* ctx  = out;                                   // (8,16,1024,64) context
    float* attw = out + (size_t)8 * 16 * 1024 * 64;      // (8,16,1024,1024) attn_w

    dim3 grid(8 * 16 * 8);   // 1024 blocks: 128 (b,h) x 8 q-chunks
    dim3 block(256);         // 8 waves (wave32)
    sdpa_wmma_kernel<<<grid, block, 0, stream>>>(Q, K, V, M, ctx, attw);
}
